// HeteroPULLModel_8160437862948
// MI455X (gfx1250) — matrix operand order repaired and checked by reading, not compiled, not run on hardware
//
#include <hip/hip_runtime.h>

// ---------------------------------------------------------------------------
// CDNA5 (gfx1250, wave32) implementation of the HGT forward pass.
// Dense GEMMs use v_wmma_f32_16x16x32_bf16 (f32 accum), float4 staging loads,
// global_prefetch of the next K tile, and compile-time epilogue variants.
// Edge phase: 3 bandwidth passes with float atomics (monotone-uint atomicMax).
// ---------------------------------------------------------------------------

typedef __attribute__((ext_vector_type(16))) __bf16 v16bf;
typedef __attribute__((ext_vector_type(8)))  float  v8f;

#define GT_M 64
#define GT_N 64
#define GT_K 32
#define LDSPAD 2

__device__ __forceinline__ float gelu_f(float x) {
  return 0.5f * x * (1.0f + erff(x * 0.7071067811865475f));
}

// C[M,N] = post( gate*(act(A)@B + bias) + (1-gate)*xprev ), bf16 WMMA compute.
// Compile-time flags: GELUA (gelu on A), RELU (post relu), SKIP (gated skip),
// BIAS (add bias). 128 threads = 4 waves; block tile 64x64; wave tile 32x32.
template <int GELUA, int RELU, int SKIP, int BIAS>
__global__ __launch_bounds__(128) void gemm_t(
    const float* __restrict__ A, const float* __restrict__ B,
    const float* __restrict__ bias, const float* __restrict__ xprev,
    const float* __restrict__ skip_gate, float* __restrict__ C,
    int M, int N, int K)
{
  __shared__ __bf16 As[GT_M][GT_K + LDSPAD];   // [m][k]
  __shared__ __bf16 Bs[GT_N][GT_K + LDSPAD];   // [n][k] (transposed stage)
  const int tid  = threadIdx.x;
  const int lane = tid & 31;
  const int wave = tid >> 5;
  const int wm = (wave >> 1) * 32;  // wave's row quadrant
  const int wn = (wave & 1) * 32;   // wave's col quadrant
  const int tileM = blockIdx.y * GT_M;
  const int tileN = blockIdx.x * GT_N;
  const int half = lane >> 4;
  const int l16  = lane & 15;

  v8f acc[2][2];
#pragma unroll
  for (int i = 0; i < 2; ++i)
#pragma unroll
    for (int j = 0; j < 2; ++j)
#pragma unroll
      for (int v = 0; v < 8; ++v) acc[i][j][v] = 0.0f;

  const int kTiles = K / GT_K;   // all K used here are multiples of 32
  for (int kb = 0; kb < kTiles; ++kb) {
    const int k0 = kb * GT_K;
    const bool more = (kb + 1) < kTiles;
    // stage A tile (64x32 f32 -> bf16) as 512 float4 chunks
#pragma unroll
    for (int r = 0; r < 4; ++r) {
      int id = tid + r * 128;
      int ml = id >> 3, kq = (id & 7) * 4;
      int grow = tileM + ml;
      float4 av = make_float4(0.f, 0.f, 0.f, 0.f);
      if (grow < M) {
        const float* ap = A + (size_t)grow * K + k0 + kq;
        av = *(const float4*)ap;
        if (more) __builtin_prefetch(ap + GT_K, 0, 1);  // global_prefetch next tile
      }
      if (GELUA) {
        av.x = gelu_f(av.x); av.y = gelu_f(av.y);
        av.z = gelu_f(av.z); av.w = gelu_f(av.w);
      }
      As[ml][kq + 0] = (__bf16)av.x;
      As[ml][kq + 1] = (__bf16)av.y;
      As[ml][kq + 2] = (__bf16)av.z;
      As[ml][kq + 3] = (__bf16)av.w;
    }
    // stage B tile (32x64 f32) transposed to [n][k] bf16
#pragma unroll
    for (int r = 0; r < 4; ++r) {
      int id = tid + r * 128;
      int kl = id >> 4, nq = (id & 15) * 4;
      int gcol = tileN + nq;
      float4 bv = make_float4(0.f, 0.f, 0.f, 0.f);
      if (gcol < N) {   // N % 4 == 0 for every GEMM in this graph
        const float* bp = B + (size_t)(k0 + kl) * N + gcol;
        bv = *(const float4*)bp;
        if (more) __builtin_prefetch(bp + (size_t)GT_K * N, 0, 1);
      }
      Bs[nq + 0][kl] = (__bf16)bv.x;
      Bs[nq + 1][kl] = (__bf16)bv.y;
      Bs[nq + 2][kl] = (__bf16)bv.z;
      Bs[nq + 3][kl] = (__bf16)bv.w;
    }
    __syncthreads();

    // Build WMMA fragments per ISA 16-bit layouts (wave32).
    v16bf af[2], bfr[2];
#pragma unroll
    for (int ti = 0; ti < 2; ++ti) {
      int m = wm + ti * 16 + l16;
#pragma unroll
      for (int i = 0; i < 16; ++i) {
        int j = i >> 1, rr = i & 1;
        int k = (j < 4) ? (2 * j + rr + 8 * half)
                        : (16 + 2 * (j - 4) + rr + 8 * half);
        af[ti][i] = As[m][k];
      }
    }
#pragma unroll
    for (int tj = 0; tj < 2; ++tj) {
      int n = wn + tj * 16 + l16;
#pragma unroll
      for (int i = 0; i < 16; ++i)
        bfr[tj][i] = Bs[n][(i < 8) ? (i + 8 * half) : (i + 8 + 8 * half)];
    }
#pragma unroll
    for (int ti = 0; ti < 2; ++ti)
#pragma unroll
      for (int tj = 0; tj < 2; ++tj)
        acc[ti][tj] = __builtin_amdgcn_wmma_f32_16x16x32_bf16(
            false, af[ti], false, bfr[tj], (short)0, acc[ti][tj], false, false);
    __syncthreads();
  }

  float g = 1.0f, og = 0.0f;
  if (SKIP) {
    float s = skip_gate[0];
    g  = 1.0f / (1.0f + expf(-s));
    og = 1.0f - g;
  }
#pragma unroll
  for (int ti = 0; ti < 2; ++ti)
#pragma unroll
    for (int tj = 0; tj < 2; ++tj)
#pragma unroll
      for (int v = 0; v < 8; ++v) {
        int row = tileM + wm + ti * 16 + v + 8 * half;   // C/D VGPR layout
        int col = tileN + wn + tj * 16 + l16;
        if (row < M && col < N) {
          float val = acc[ti][tj][v];
          if (BIAS) val += bias[col];
          if (SKIP) val = g * val + og * xprev[(size_t)row * N + col];
          if (RELU) val = fmaxf(val, 0.0f);
          C[(size_t)row * N + col] = val;
        }
      }
}

// ---------------------------------------------------------------------------
__global__ void fill_u32(unsigned* __restrict__ p, unsigned v, int n) {
  int t = blockIdx.x * blockDim.x + threadIdx.x;
  if (t < n) p[t] = v;
}

__device__ __forceinline__ unsigned mono_f32(float f) {
  unsigned b = __float_as_uint(f);
  return (b & 0x80000000u) ? ~b : (b | 0x80000000u);
}
__device__ __forceinline__ float demono_f32(unsigned u) {
  return (u & 0x80000000u) ? __uint_as_float(u & 0x7fffffffu)
                           : __uint_as_float(~u);
}

// rel [4][32][32] -> block-diagonal [128][128] so einsum('nhd,hde->nhe') == GEMM
__global__ void expand_blockdiag(const float* __restrict__ rel, float* __restrict__ Bd) {
  int t = blockIdx.x * blockDim.x + threadIdx.x;
  if (t >= 128 * 128) return;
  int kk = t >> 7, nn = t & 127;
  float v = 0.0f;
  if ((kk >> 5) == (nn >> 5))
    v = rel[(kk >> 5) * 1024 + (kk & 31) * 32 + (nn & 31)];
  Bd[t] = v;
}

// pass 1: per-(edge,head) attention logit + segment max (monotone uint atomicMax)
__global__ void edge_logits_max(const float* __restrict__ qdst,
                                const float* __restrict__ krel,
                                const int* __restrict__ ei,
                                const float* __restrict__ p_rel,
                                float* __restrict__ Lbuf,
                                unsigned* __restrict__ mbuf, int E) {
  int t = blockIdx.x * blockDim.x + threadIdx.x;
  if (t >= E * 4) return;
  int e = t >> 2, h = t & 3;
  int src = ei[e], dst = ei[E + e];
  const float4* q4 = (const float4*)(qdst + (size_t)dst * 128 + h * 32);
  const float4* k4 = (const float4*)(krel + (size_t)src * 128 + h * 32);
  float acc = 0.0f;
#pragma unroll
  for (int i = 0; i < 8; ++i) {
    float4 a = q4[i], b = k4[i];
    acc += a.x * b.x + a.y * b.y + a.z * b.z + a.w * b.w;
  }
  float L = acc * p_rel[h] * 0.1767766952966369f;   // 1/sqrt(32)
  Lbuf[t] = L;
  atomicMax(&mbuf[(size_t)dst * 4 + h], mono_f32(L));
}

// pass 2: ex = exp(L - m); Lbuf <- ex; segment sum
__global__ void edge_expsum(const int* __restrict__ ei,
                            const unsigned* __restrict__ mbuf,
                            float* __restrict__ Lbuf,
                            float* __restrict__ sbuf, int E) {
  int t = blockIdx.x * blockDim.x + threadIdx.x;
  if (t >= E * 4) return;
  int e = t >> 2, h = t & 3;
  int dst = ei[E + e];
  float m = demono_f32(mbuf[(size_t)dst * 4 + h]);
  if (!(m > -3.0e38f) || m != m) m = 0.0f;          // mirror where(isfinite(m),m,0)
  float ex = expf(Lbuf[t] - m);
  Lbuf[t] = ex;
  atomicAdd(&sbuf[(size_t)dst * 4 + h], ex);
}

// pass 3: agg[dst,:] += alpha * v_rel[src,:]
__global__ void edge_scatter(const int* __restrict__ ei,
                             const float* __restrict__ vrel,
                             const float* __restrict__ Lbuf,
                             const float* __restrict__ sbuf,
                             float* __restrict__ agg, int E) {
  int t = blockIdx.x * blockDim.x + threadIdx.x;
  if (t >= E * 128) return;
  int e = t >> 7, d = t & 127, h = d >> 5;
  int src = ei[e], dst = ei[E + e];
  float alpha = Lbuf[e * 4 + h] / (sbuf[(size_t)dst * 4 + h] + 1e-16f);
  atomicAdd(&agg[(size_t)dst * 128 + d], alpha * vrel[(size_t)src * 128 + d]);
}

// final: clip(<z_C[q0], z_D[q1]>, -10, 10)
__global__ void edge_score(const float* __restrict__ zC, const float* __restrict__ zD,
                           const int* __restrict__ eli, float* __restrict__ out, int NQ) {
  int t = blockIdx.x * blockDim.x + threadIdx.x;
  if (t >= NQ) return;
  int c = eli[t], d = eli[NQ + t];
  const float4* a = (const float4*)(zC + (size_t)c * 64);
  const float4* b = (const float4*)(zD + (size_t)d * 64);
  float acc = 0.0f;
#pragma unroll
  for (int i = 0; i < 16; ++i) {
    float4 x = a[i], y = b[i];
    acc += x.x * y.x + x.y * y.y + x.z * y.z + x.w * y.w;
  }
  out[t] = fminf(fmaxf(acc, -10.0f), 10.0f);
}

// ---------------------------------------------------------------------------
extern "C" void kernel_launch(void* const* d_in, const int* in_sizes, int n_in,
                              void* d_out, int out_size, void* d_ws, size_t ws_size,
                              hipStream_t stream)
{
  (void)in_sizes; (void)n_in; (void)out_size; (void)ws_size;
  auto F  = [&](int i) { return (const float*)d_in[i]; };
  auto Ii = [&](int i) { return (const int*)d_in[i]; };

  // ---- input index map (setup_inputs insertion order; params = sorted pytree)
  // 0:x_Compound 1:x_Disease
  // conv block leaf offsets (sorted keys): a_rel+0..5 (CbG,CtD,DaG,DtC,GaD,GbC),
  // k_b+6..8, k_w+9..11, m_rel+12..17, out_b+18..20, out_w+21..23,
  // p_rel+24..29, q_b+30..32, q_w+33..35, skip+36..38, v_b+39..41, v_w+42..44.
  // conv0 at 2, conv1 at 47. 92:gene_embed 93/94:in_b{C,D} 95/96:in_w{C,D}
  // 97..99:lin_b{C,D,G} 100..102:lin_w{C,D,G}
  // 103:ei_CtD 104:ei_DtC 105:ei_CbG 106:ei_GbC 107:ei_GaD 108:ei_DaG 109:eli
  const int Nt[3] = {10000, 4000, 20000};   // Compound, Disease, Gene

  // ---- workspace carve-out (256B aligned)
  char* w = (char*)d_ws;
  auto alloc = [&](size_t bytes) -> void* {
    void* p = (void*)w; w += (bytes + 255) & ~(size_t)255; return p;
  };
  float* xC  = (float*)alloc((size_t)10000 * 128 * 4);
  float* xD  = (float*)alloc((size_t)4000  * 128 * 4);
  float* xG2 = (float*)alloc((size_t)20000 * 128 * 4);
  float* yC  = (float*)alloc((size_t)10000 * 128 * 4);
  float* yD  = (float*)alloc((size_t)4000  * 128 * 4);
  float* yG  = (float*)alloc((size_t)20000 * 128 * 4);
  float* kb3[3]; float* qb3[3]; float* vb3[3];
  for (int t = 0; t < 3; ++t) {
    kb3[t] = (float*)alloc((size_t)Nt[t] * 128 * 4);
    qb3[t] = (float*)alloc((size_t)Nt[t] * 128 * 4);
    vb3[t] = (float*)alloc((size_t)Nt[t] * 128 * 4);
  }
  float* Bd   = (float*)alloc(128 * 128 * 4);
  float* Bd2  = (float*)alloc(128 * 128 * 4);
  float* krel[2]; float* vrel[2];
  for (int r = 0; r < 2; ++r) {
    krel[r] = (float*)alloc((size_t)20000 * 128 * 4);
    vrel[r] = (float*)alloc((size_t)20000 * 128 * 4);
  }
  float*    Lbuf = (float*)alloc((size_t)160000 * 4 * 4);
  unsigned* mbuf = (unsigned*)alloc((size_t)20000 * 4 * 4);
  float*    sbuf = (float*)alloc((size_t)20000 * 4 * 4);
  float*    agg  = (float*)alloc((size_t)20000 * 128 * 4);
  float*    zC   = (float*)alloc((size_t)10000 * 64 * 4);
  float*    zD   = (float*)alloc((size_t)4000  * 64 * 4);

  // dispatch to the 4 compiled epilogue variants
  auto gemm = [&](const float* A, const float* B, const float* bias,
                  const float* xprev, const float* gate, float* Co,
                  int M, int N, int K, int geluA, int postRelu) {
    dim3 grid((N + GT_N - 1) / GT_N, (M + GT_M - 1) / GT_M);
    dim3 blk(128);
    if (geluA)           gemm_t<1, 1, 1, 1><<<grid, blk, 0, stream>>>(A, B, bias, xprev, gate, Co, M, N, K);
    else if (!bias)      gemm_t<0, 0, 0, 0><<<grid, blk, 0, stream>>>(A, B, bias, xprev, gate, Co, M, N, K);
    else if (postRelu)   gemm_t<0, 1, 0, 1><<<grid, blk, 0, stream>>>(A, B, bias, xprev, gate, Co, M, N, K);
    else                 gemm_t<0, 0, 0, 1><<<grid, blk, 0, stream>>>(A, B, bias, xprev, gate, Co, M, N, K);
  };
  auto fill = [&](void* p, unsigned v, int n) {
    fill_u32<<<dim3((n + 255) / 256), dim3(256), 0, stream>>>((unsigned*)p, v, n);
  };

  // ---- input projections (ReLU)
  gemm(F(0), F(95), F(93), nullptr, nullptr, xC, 10000, 128, 2048, 0, 1);
  gemm(F(1), F(96), F(94), nullptr, nullptr, xD, 4000, 128, 4000, 0, 1);

  const float* xin[3] = { xC, xD, F(92) };     // gene_embed read in place
  float* xout[3] = { yC, yD, yG };

  struct RelIn { int rel; int src; int ei; int E; };
  const RelIn relTab[3][2] = {
    { {3, 1, 104,  50000}, {5, 2, 106, 100000} },  // dst Compound: DtC, GbC
    { {1, 0, 103,  50000}, {4, 2, 107,  60000} },  // dst Disease : CtD, GaD
    { {0, 0, 105, 100000}, {2, 1, 108,  60000} },  // dst Gene    : CbG, DaG
  };

  for (int l = 0; l < 2; ++l) {
    const int pb = (l == 0) ? 2 : 47;
    for (int t = 0; t < 3; ++t) {
      gemm(xin[t], F(pb +  9 + t), F(pb +  6 + t), nullptr, nullptr, kb3[t], Nt[t], 128, 128, 0, 0);
      gemm(xin[t], F(pb + 33 + t), F(pb + 30 + t), nullptr, nullptr, qb3[t], Nt[t], 128, 128, 0, 0);
      gemm(xin[t], F(pb + 42 + t), F(pb + 39 + t), nullptr, nullptr, vb3[t], Nt[t], 128, 128, 0, 0);
    }
    for (int dt = 0; dt < 3; ++dt) {
      const int n = Nt[dt];
      fill(mbuf, 0x007FFFFFu, n * 4);      // mono(-inf)
      fill(sbuf, 0u, n * 4);
      fill(agg,  0u, n * 128);
      size_t Loff[2] = { 0, (size_t)relTab[dt][0].E * 4 };
      for (int ri = 0; ri < 2; ++ri) {
        const RelIn& R = relTab[dt][ri];
        expand_blockdiag<<<64, 256, 0, stream>>>(F(pb +  0 + R.rel), Bd);
        gemm(kb3[R.src], Bd,  nullptr, nullptr, nullptr, krel[ri], Nt[R.src], 128, 128, 0, 0);
        expand_blockdiag<<<64, 256, 0, stream>>>(F(pb + 12 + R.rel), Bd2);
        gemm(vb3[R.src], Bd2, nullptr, nullptr, nullptr, vrel[ri], Nt[R.src], 128, 128, 0, 0);
      }
      for (int ri = 0; ri < 2; ++ri) {      // logits + max over ALL incoming rels
        const RelIn& R = relTab[dt][ri];
        int th = R.E * 4;
        edge_logits_max<<<(th + 255) / 256, 256, 0, stream>>>(
            qb3[dt], krel[ri], Ii(R.ei), F(pb + 24 + R.rel), Lbuf + Loff[ri], mbuf, R.E);
      }
      for (int ri = 0; ri < 2; ++ri) {      // exp + segment sum
        const RelIn& R = relTab[dt][ri];
        int th = R.E * 4;
        edge_expsum<<<(th + 255) / 256, 256, 0, stream>>>(
            Ii(R.ei), mbuf, Lbuf + Loff[ri], sbuf, R.E);
      }
      for (int ri = 0; ri < 2; ++ri) {      // normalized scatter of V
        const RelIn& R = relTab[dt][ri];
        int th = R.E * 128;
        edge_scatter<<<(th + 255) / 256, 256, 0, stream>>>(
            Ii(R.ei), vrel[ri], Lbuf + Loff[ri], sbuf, agg, R.E);
      }
      // out proj: gelu(agg)@out_w + out_b, sigmoid-gated skip, ReLU
      gemm(agg, F(pb + 21 + dt), F(pb + 18 + dt), xin[dt], F(pb + 36 + dt),
           xout[dt], n, 128, 128, 1, 1);
    }
    xin[0] = yC; xin[1] = yD; xin[2] = yG;
    xout[0] = xC; xout[1] = xD; xout[2] = xG2;
  }

  // ---- final linears (Gene unused by the scorer) + link scoring
  gemm(xC, F(100), F(97), nullptr, nullptr, zC, 10000, 64, 128, 0, 0);
  gemm(xD, F(101), F(98), nullptr, nullptr, zD, 4000, 64, 128, 0, 0);
  edge_score<<<(100000 + 255) / 256, 256, 0, stream>>>(
      zC, zD, Ii(109), (float*)d_out, 100000);
}
